// ContrastiveLoss_77369540870625
// MI455X (gfx1250) — compile-verified, matrix operand order checked by reading, not run on hardware
//
#include <hip/hip_runtime.h>
#include <hip/hip_bf16.h>

typedef __attribute__((ext_vector_type(16))) _Float16 v16h;
typedef __attribute__((ext_vector_type(8)))  _Float16 v8h;
typedef __attribute__((ext_vector_type(8)))  float    v8f;
typedef int v4i __attribute__((vector_size(16)));   // matches async-LDS builtin param type

#define CL_B 4
#define CL_P 4096
#define CL_H 256
#define CL_TEMP_INV (1.0f / 0.07f)
#define CL_NEG_INF (-1.0e30f)
#define CL_TILES_PER_BATCH (CL_P / 16)          // 256 tiles of 16 per batch
#define CL_NTILES (CL_B * CL_TILES_PER_BATCH)   // 1024 wave tiles total

// ---- CDNA5 async global->LDS copy path (ASYNCcnt), with safe fallback ----
#if defined(__AMDGCN__) && __has_builtin(__builtin_amdgcn_global_load_async_to_lds_b128)
#define CL_HAVE_ASYNC 1
#else
#define CL_HAVE_ASYNC 0
#endif

template <int N>
__device__ __forceinline__ void wait_async() {
#if CL_HAVE_ASYNC
#if __has_builtin(__builtin_amdgcn_s_wait_asynccnt)
    __builtin_amdgcn_s_wait_asynccnt(N);
#else
    asm volatile("s_wait_asynccnt %0" :: "i"(N) : "memory");
#endif
#endif
}

// Copy 16 bytes global -> LDS without touching the VGPR data path (async DMA).
__device__ __forceinline__ void copy16_g2l(const char* g, char* l) {
#if CL_HAVE_ASYNC
    __builtin_amdgcn_global_load_async_to_lds_b128(
        (__attribute__((address_space(1))) v4i*)g,
        (__attribute__((address_space(3))) v4i*)l, 0, 0);
#else
    *(v8h*)l = *(const v8h*)g;   // synchronous fallback
#endif
}

// ---------- shuffle reduction helpers (wave32; xor masks < 16 stay in a 16-lane half) ----------
__device__ __forceinline__ float half16_max(float v) {
    v = fmaxf(v, __shfl_xor(v, 1, 32));
    v = fmaxf(v, __shfl_xor(v, 2, 32));
    v = fmaxf(v, __shfl_xor(v, 4, 32));
    v = fmaxf(v, __shfl_xor(v, 8, 32));
    return v;
}
__device__ __forceinline__ float half16_sum(float v) {
    v += __shfl_xor(v, 1, 32);
    v += __shfl_xor(v, 2, 32);
    v += __shfl_xor(v, 4, 32);
    v += __shfl_xor(v, 8, 32);
    return v;
}
__device__ __forceinline__ float wave_sum(float v) {
    v += __shfl_xor(v, 1, 32);
    v += __shfl_xor(v, 2, 32);
    v += __shfl_xor(v, 4, 32);
    v += __shfl_xor(v, 8, 32);
    v += __shfl_xor(v, 16, 32);
    return v;
}

// ---------- kernel 1: label masks + batch_ok ----------
__global__ void labels_kernel(const int* __restrict__ labels,
                              float* __restrict__ negBias,
                              int* __restrict__ posOk) {
    const int b = blockIdx.x;
    __shared__ int sValid, sPos, sNeg, sOk;
    if (threadIdx.x == 0) { sValid = 0; sPos = 0; sNeg = 0; }
    __syncthreads();
    int v = 0, pp = 0, nn = 0;
    for (int p = threadIdx.x; p < CL_P; p += blockDim.x) {
        int L = labels[b * CL_P + p];
        if (L != -100) { v++; if (L == 1) pp++; else nn++; }
    }
    atomicAdd(&sValid, v); atomicAdd(&sPos, pp); atomicAdd(&sNeg, nn);
    __syncthreads();
    if (threadIdx.x == 0) sOk = (sValid >= 2 && sPos > 0 && sNeg > 0) ? 1 : 0;
    __syncthreads();
    const int ok = sOk;
    for (int p = threadIdx.x; p < CL_P; p += blockDim.x) {
        int L = labels[b * CL_P + p];
        negBias[b * CL_P + p] = (L != -100 && L != 1) ? 0.0f : CL_NEG_INF;
        posOk[b * CL_P + p]   = (L == 1 && ok) ? 1 : 0;
    }
}

// ---------- kernel 2: L2-normalize, fold 1/T into g, convert to f16 ----------
__global__ __launch_bounds__(256) void normalize_kernel(
        const float* __restrict__ gin, const float* __restrict__ ein,
        _Float16* __restrict__ gh, _Float16* __restrict__ eh) {
    const int lane = threadIdx.x & 31;
    const int wave = threadIdx.x >> 5;
    const int NR = CL_B * CL_P;
    int r = blockIdx.x * 8 + wave;            // 0 .. 2*NR-1
    const bool isE = r >= NR;
    const int row = isE ? (r - NR) : r;
    const float* src = (isE ? ein : gin) + (size_t)row * CL_H;
    float vals[8];
    float ss = 0.0f;
#pragma unroll
    for (int i = 0; i < 8; ++i) {
        vals[i] = src[lane * 8 + i];
        ss += vals[i] * vals[i];
    }
#pragma unroll
    for (int m = 1; m < 32; m <<= 1) ss += __shfl_xor(ss, m, 32);
    float nrm = fmaxf(sqrtf(ss), 1e-12f);     // F.normalize: clamp norm at 1e-12
    float scale = (isE ? 1.0f : CL_TEMP_INV) / nrm;
    _Float16* dst = (isE ? eh : gh) + (size_t)row * CL_H;
#pragma unroll
    for (int i = 0; i < 8; ++i) dst[lane * 8 + i] = (_Float16)(vals[i] * scale);
}

// Load a 16x32 f16 WMMA fragment for the row this lane owns.
// 16-bit fragment layout (ISA 7.12.2): lanes 0-15 hold K=0..7 and 16..23,
// lanes 16-31 hold K=8..15 and 24..31 -> two contiguous 16B chunks per lane.
__device__ __forceinline__ v16h load_frag(const _Float16* __restrict__ rowPtr,
                                          int ks, int kOff) {
    const _Float16* p = rowPtr + ks * 32 + kOff;
    v8h c0 = *(const v8h*)(p);
    v8h c1 = *(const v8h*)(p + 16);
    v16h f;
#pragma unroll
    for (int i = 0; i < 8; ++i) { f[i] = c0[i]; f[i + 8] = c1[i]; }
    return f;
}

// Stage one 16x256 f16 B-tile (8 KB) into LDS; 256 threads x 32 B each.
__device__ __forceinline__ void stage_tile(const _Float16* __restrict__ ehBatch,
                                           int tn, _Float16* ldsBuf, int tid) {
    const int byteOff = tid * 32;
    const int rowL = byteOff >> 9;            // 512 B per row
    const int colB = byteOff & 511;
    const char* g = (const char*)(ehBatch + (size_t)(tn * 16 + rowL) * CL_H) + colB;
    char* l = (char*)ldsBuf + rowL * 512 + colB;
    copy16_g2l(g, l);
    copy16_g2l(g + 16, l + 16);
}

// ---------- kernel 3: fused GEMM + online logsumexp over neg columns ----------
// Block = 8 waves = 128 rows of one batch; B tiles staged to LDS once per
// block (double-buffered, async DMA) and shared by all 8 waves.
__global__ __launch_bounds__(256) void gemm_lse_kernel(
        const _Float16* __restrict__ gh, const _Float16* __restrict__ eh,
        const float* __restrict__ negBias, const int* __restrict__ posOk,
        float* __restrict__ partialT, float* __restrict__ partialC) {
    __shared__ _Float16 ldsB[2][16 * CL_H];   // 2 x 8 KB double buffer

    const int tid  = threadIdx.x;
    const int lane = tid & 31;
    const int wave = tid >> 5;
    const int t  = blockIdx.x * 8 + wave;        // wave tile id, 0..1023
    const int b  = t >> 8;                       // batch (uniform per block)
    const int tm = t & (CL_TILES_PER_BATCH - 1); // M-tile within batch
    const int row0 = tm * 16;
    const int lh = lane & 15;
    const bool hiHalf = lane >= 16;
    const int kOff = hiHalf ? 8 : 0;

    const _Float16* ehBatch = eh + (size_t)b * CL_P * CL_H;

    // A fragments (16 rows x 256 K) stay resident: 8 x v16h = 64 VGPRs.
    const _Float16* gRow = gh + ((size_t)b * CL_P + row0 + lh) * CL_H;
    v16h afrag[8];
#pragma unroll
    for (int ks = 0; ks < 8; ++ks) afrag[ks] = load_frag(gRow, ks, kOff);

    // Online softmax state per row. C layout: VGPR r holds row r (lanes 0-15)
    // and row r+8 (lanes 16-31), N = lane&15. State replicated within a half.
    float m_run[8], s_run[8], diagv[8];
#pragma unroll
    for (int r = 0; r < 8; ++r) { m_run[r] = CL_NEG_INF; s_run[r] = 0.0f; diagv[r] = 0.0f; }

    stage_tile(ehBatch, 0, ldsB[0], tid);     // prologue: tile 0 in flight

    for (int tn = 0; tn < CL_TILES_PER_BATCH; ++tn) {
        const int cur = tn & 1;
        if (tn + 1 < CL_TILES_PER_BATCH) {
            stage_tile(ehBatch, tn + 1, ldsB[cur ^ 1], tid);  // prefetch next
            wait_async<2>();    // only the just-issued stage may remain in flight
        } else {
            wait_async<0>();
        }
        __syncthreads();        // ldsB[cur] complete block-wide

        const int col = tn * 16 + lh;
        const float bias = negBias[(size_t)b * CL_P + col];   // overlaps LDS reads

        // Preload ALL 8 B fragments into distinct registers (16 ds_load_b128
        // in flight, one dscnt wait), then chain the 8 WMMAs back-to-back.
        const _Float16* bRow = &ldsB[cur][lh * CL_H];
        v16h bfrag[8];
#pragma unroll
        for (int ks = 0; ks < 8; ++ks) bfrag[ks] = load_frag(bRow, ks, kOff);

        v8f c = {};
#pragma unroll
        for (int ks = 0; ks < 8; ++ks) {
            c = __builtin_amdgcn_wmma_f32_16x16x32_f16(
                    false, afrag[ks], false, bfrag[ks], (short)0, c, false, false);
        }

        if (tn == tm) {   // diagonal block: capture sim[i,i] (unmasked)
#pragma unroll
            for (int r = 0; r < 8; ++r) {
                int rowInTile = r + (hiHalf ? 8 : 0);
                float d = (rowInTile == lh) ? c[r] : 0.0f;
                diagv[r] = half16_sum(d);   // exactly one lane contributes
            }
        }
#pragma unroll
        for (int r = 0; r < 8; ++r) {
            float masked = c[r] + bias;                 // ~NEG_INF where not a neg column
            float tmax = half16_max(masked);
            float newm = fmaxf(m_run[r], tmax);
            if (newm > -1e29f) {                        // uniform within the 16-lane half
                float pe   = __expf(masked - newm);
                float rsum = half16_sum(pe);
                s_run[r] = s_run[r] * __expf(m_run[r] - newm) + rsum;
                m_run[r] = newm;
            }
        }
        __syncthreads();  // all waves done with ldsB[cur] before it is re-staged
    }

    // lse_i = log(exp(diag) + sum_neg exp) ; loss_i = lse_i - diag_i
    float localT = 0.0f, localC = 0.0f;
#pragma unroll
    for (int r = 0; r < 8; ++r) {
        int row = row0 + r + (hiHalf ? 8 : 0);
        float md  = fmaxf(m_run[r], diagv[r]);
        float sum = __expf(diagv[r] - md) + s_run[r] * __expf(m_run[r] - md);
        float loss = md + __logf(sum) - diagv[r];
        if (lh == 0 && posOk[(size_t)b * CL_P + row]) { localT += loss; localC += 1.0f; }
    }
    float wT = wave_sum(localT);
    float wC = wave_sum(localC);
    if (lane == 0) { partialT[t] = wT; partialC[t] = wC; }
}

// ---------- kernel 4: deterministic final reduction ----------
__global__ void finalize_kernel(const float* __restrict__ partialT,
                                const float* __restrict__ partialC,
                                float* __restrict__ out) {
    __shared__ float sT[256], sC[256];
    const int tid = threadIdx.x;
    float tt = 0.0f, cc = 0.0f;
#pragma unroll
    for (int i = 0; i < 4; ++i) {
        tt += partialT[tid * 4 + i];
        cc += partialC[tid * 4 + i];
    }
    sT[tid] = tt; sC[tid] = cc;
    __syncthreads();
    for (int s = 128; s > 0; s >>= 1) {
        if (tid < s) { sT[tid] += sT[tid + s]; sC[tid] += sC[tid + s]; }
        __syncthreads();
    }
    if (tid == 0) out[0] = (sC[0] == 0.0f) ? 0.0f : (sT[0] / sC[0]);
}

extern "C" void kernel_launch(void* const* d_in, const int* in_sizes, int n_in,
                              void* d_out, int out_size, void* d_ws, size_t ws_size,
                              hipStream_t stream) {
    const float* greek   = (const float*)d_in[0];   // [B,P,H] f32
    const float* english = (const float*)d_in[1];   // [B,P,H] f32
    const int*   labels  = (const int*)d_in[2];     // [B,P] i32
    float* out = (float*)d_out;

    // Workspace layout (~16.9 MB total)
    char* ws = (char*)d_ws;
    size_t off = 0;
    _Float16* gh = (_Float16*)(ws + off); off += (size_t)CL_B * CL_P * CL_H * sizeof(_Float16);
    _Float16* eh = (_Float16*)(ws + off); off += (size_t)CL_B * CL_P * CL_H * sizeof(_Float16);
    float* negBias = (float*)(ws + off);  off += (size_t)CL_B * CL_P * sizeof(float);
    int*   posOk   = (int*)(ws + off);    off += (size_t)CL_B * CL_P * sizeof(int);
    float* partialT = (float*)(ws + off); off += CL_NTILES * sizeof(float);
    float* partialC = (float*)(ws + off); off += CL_NTILES * sizeof(float);
    (void)ws_size; (void)in_sizes; (void)n_in; (void)out_size;

    labels_kernel<<<CL_B, 256, 0, stream>>>(labels, negBias, posOk);

    // 2*B*P rows, 8 waves (rows) per 256-thread block
    normalize_kernel<<<(2 * CL_B * CL_P) / 8, 256, 0, stream>>>(greek, english, gh, eh);

    // 1024 wave tiles, 8 waves per block -> 128 blocks
    gemm_lse_kernel<<<CL_NTILES / 8, 256, 0, stream>>>(gh, eh, negBias, posOk,
                                                       partialT, partialC);

    finalize_kernel<<<1, 256, 0, stream>>>(partialT, partialC, out);
}
